// FNOBlock_48962627175213
// MI455X (gfx1250) — compile-verified, matrix operand order checked by reading, not run on hardware
//
#include <hip/hip_runtime.h>

// ---------------------------------------------------------------------------
// FNO block for MI455X (gfx1250, wave32, WMMA), software-pipelined.
//   k_tables  : pre-swizzled f16 WMMA B-fragments for 64-mode DFT (Ffrag) and
//               irfft synthesis (Gfrag)
//   k_convert : x f32 -> f16 row-major
//   k_mlp     : phi_p/phi_n, gamma/beta from emb
//   k_dft     : (2048 x 8192) @ (8192 x 128) via v_wmma_f32_16x16x32_f16,
//               double-buffered fragments, K split into 8 chunks
//   k_mix     : reduce chunks, complex channel-mix, * phi -> fused A cols 0..127
//   k_lin     : A cols 128..191 = (1+gamma)*lin_w; bias = (1+gamma)*lin_b+beta
//   k_final   : per-b (64 x 192) @ (192 x 8192) WMMA GEMM, double-buffered,
//               epilogue +bias, SiLU -> d_out (f32)
// ---------------------------------------------------------------------------

typedef __attribute__((ext_vector_type(16))) _Float16 v16h;
typedef __attribute__((ext_vector_type(8)))  _Float16 v8h;
typedef __attribute__((ext_vector_type(8)))  float    v8f;

union AV { v16h v; v8h h8[2]; };

#define DEV static __device__ __forceinline__

constexpr int BB = 32, CC = 64, LL = 8192, MM = 32;
constexpr int ROWS   = BB * CC;         // 2048 flattened (b,c) rows
constexpr int NCOEF  = 4 * MM;          // 128 coefficient slots
constexpr int KF     = NCOEF + CC;      // 192 fused-K for final GEMM
constexpr int NEGB   = LL / 2 + 1 - MM; // 4065 first negative-block mode
constexpr int KCH    = 8;               // K chunks in forward DFT
constexpr int NTILES = LL / 16;         // 512

constexpr float INVS = 0.011048543456039806f;  // 1/sqrt(8192), ortho norm
constexpr float WANG = 7.6699039394282058e-4f; // 2*pi/8192

DEV int kmap(int h, int laneHi) {
  return (h < 8) ? ((laneHi ? 8 : 0) + h) : ((laneHi ? 24 : 16) + (h - 8));
}
DEV float sigmoidf(float v) { return 1.0f / (1.0f + __expf(-v)); }

DEV v8f wmma16(v16h a, v16h b, v8f c) {
  return __builtin_amdgcn_wmma_f32_16x16x32_f16(false, a, false, b, (short)0, c,
                                                false, false);
}

// ---------------------------------------------------------------- tables ----
__global__ void k_tables(_Float16* __restrict__ ffrag, _Float16* __restrict__ gfrag) {
  int t = blockIdx.x * blockDim.x + threadIdx.x;
  if (t < (1 << 20)) {
    // Ffrag layout: [coltile(8)][kiter(256)][lane(32)][half(16)]
    int f = t;
    int h = f & 15, lane = (f >> 4) & 31, kit = (f >> 9) & 255, ct = f >> 17;
    int laneHi = lane >> 4;
    int col = ct * 16 + (lane & 15);
    int l   = kit * 32 + kmap(h, laneHi);
    int m; bool im;
    if (col < 32)       { m = col;             im = false; }
    else if (col < 64)  { m = col - 32;        im = true;  }
    else if (col < 96)  { m = NEGB + col - 64; im = false; }
    else                { m = NEGB + col - 96; im = true;  }
    float s, c;
    __sincosf(WANG * (float)((m * l) & (LL - 1)), &s, &c);
    ffrag[f] = (_Float16)((im ? -s : c) * INVS);
  } else {
    // Gfrag layout: [ntile(512)][kblk(4)][lane(32)][half(16)]
    int g = t - (1 << 20);
    int h = g & 15, lane = (g >> 4) & 31, kk = (g >> 9) & 3, nt = g >> 11;
    int laneHi = lane >> 4;
    int l   = nt * 16 + (lane & 15);
    int row = kk * 32 + kmap(h, laneHi);
    float val, s, c;
    if (row < 32) {                        // Re, positive modes
      int m = row; float a = (m == 0) ? 1.0f : 2.0f;
      __sincosf(WANG * (float)((m * l) & (LL - 1)), &s, &c);
      val = a * c * INVS;
    } else if (row < 64) {                 // Im, positive modes (DC imag ignored)
      int m = row - 32;
      __sincosf(WANG * (float)((m * l) & (LL - 1)), &s, &c);
      val = (m == 0) ? 0.0f : -2.0f * s * INVS;
    } else if (row < 96) {                 // Re, negative-block modes
      int j = row - 64; int m = NEGB + j;
      float a = (j == 31) ? 1.0f : 2.0f;   // j==31 is Nyquist
      __sincosf(WANG * (float)((m * l) & (LL - 1)), &s, &c);
      val = a * c * INVS;
    } else {                               // Im, negative-block (Nyquist imag ignored)
      int j = row - 96; int m = NEGB + j;
      __sincosf(WANG * (float)((m * l) & (LL - 1)), &s, &c);
      val = (j == 31) ? 0.0f : -2.0f * s * INVS;
    }
    gfrag[g] = (_Float16)val;
  }
}

// --------------------------------------------------------------- convert ----
__global__ void k_convert(const float* __restrict__ x, _Float16* __restrict__ x16) {
  size_t i = ((size_t)blockIdx.x * blockDim.x + threadIdx.x) * 8;
  float4 a = *(const float4*)(x + i);
  float4 b = *(const float4*)(x + i + 4);
  v8h h;
  h[0] = (_Float16)a.x; h[1] = (_Float16)a.y; h[2] = (_Float16)a.z; h[3] = (_Float16)a.w;
  h[4] = (_Float16)b.x; h[5] = (_Float16)b.y; h[6] = (_Float16)b.z; h[7] = (_Float16)b.w;
  *(v8h*)(x16 + i) = h;
}

// ------------------------------------------------------------------- mlp ----
__global__ void k_mlp(const float* __restrict__ emb,
                      const float* __restrict__ Arp, const float* __restrict__ Aip,
                      const float* __restrict__ Arn, const float* __restrict__ Ain,
                      const float* __restrict__ w1, const float* __restrict__ b1,
                      const float* __restrict__ w2, const float* __restrict__ b2,
                      float* __restrict__ phi_p, float* __restrict__ phi_n,
                      float* __restrict__ gamma, float* __restrict__ beta) {
  int b = blockIdx.x, t = threadIdx.x; // 64 threads
  __shared__ float se[256];
  __shared__ float sh[64];
  for (int e = t; e < 256; e += 64) se[e] = emb[b * 256 + e];
  __syncthreads();
  float acc = b1[t];
  for (int e = 0; e < 256; ++e) acc += se[e] * w1[t * 256 + e];
  sh[t] = acc * sigmoidf(acc);
  __syncthreads();
  float g = b2[t], be = b2[64 + t];
  for (int j = 0; j < 64; ++j) {
    float hv = sh[j];
    g  += hv * w2[t * 64 + j];
    be += hv * w2[(64 + t) * 64 + j];
  }
  gamma[b * 64 + t] = g;
  beta[b * 64 + t]  = be;
  if (t < 32) {
    float rp = 0, ip = 0, rn = 0, in_ = 0;
    for (int e = 0; e < 256; ++e) {
      float ev = se[e];
      rp += ev * Arp[t * 256 + e]; ip += ev * Aip[t * 256 + e];
      rn += ev * Arn[t * 256 + e]; in_ += ev * Ain[t * 256 + e];
    }
    phi_p[(b * 32 + t) * 2 + 0] = rp; phi_p[(b * 32 + t) * 2 + 1] = ip;
    phi_n[(b * 32 + t) * 2 + 0] = rn; phi_n[(b * 32 + t) * 2 + 1] = in_;
  }
}

// ----------------------------------------------------------- forward DFT ----
// grid (128 row-tiles, 8 k-chunks), block 64 (2 waves; wave w -> cols 64w..64w+63)
// Double-buffered: loads for k-step i+1 issued before WMMAs of step i, so the
// hardware can overlap VMEM latency with the matrix pipe (loads return in
// order -> nonzero loadcnt waits instead of s_wait_loadcnt 0 per WMMA).
__global__ void k_dft(const _Float16* __restrict__ x16,
                      const _Float16* __restrict__ ffrag,
                      float* __restrict__ partial) {
  int lane = threadIdx.x & 31;
  int wv   = threadIdx.x >> 5;
  int laneHi = lane >> 4;
  int rowt = blockIdx.x, kch = blockIdx.y;
  int row = rowt * 16 + (lane & 15);
  const _Float16* xr = x16 + (size_t)row * LL + (size_t)kch * 1024;
  // Ffrag element: ((ct_global*256 + kch*32 + ki)*32 + lane)*16
  const _Float16* fb = ffrag + (size_t)(wv * 4) * 131072
                             + (size_t)(kch * 32) * 512 + (size_t)lane * 16;

  v8f acc[4] = {};
  AV a_cur; v16h b_cur[4];
  a_cur.h8[0] = *(const v8h*)(xr + (laneHi << 3));
  a_cur.h8[1] = *(const v8h*)(xr + 16 + (laneHi << 3));
#pragma unroll
  for (int ct = 0; ct < 4; ++ct)
    b_cur[ct] = *(const v16h*)(fb + (size_t)ct * 131072);

  for (int ki = 0; ki < 31; ++ki) {
    int kb = (ki + 1) * 32;
    if (ki < 24) __builtin_prefetch(xr + kb + 256, 0, 3); // global_prefetch_b8
    AV a_nxt; v16h b_nxt[4];
    a_nxt.h8[0] = *(const v8h*)(xr + kb + (laneHi << 3));
    a_nxt.h8[1] = *(const v8h*)(xr + kb + 16 + (laneHi << 3));
    const _Float16* fk = fb + (size_t)(ki + 1) * 512;
#pragma unroll
    for (int ct = 0; ct < 4; ++ct)
      b_nxt[ct] = *(const v16h*)(fk + (size_t)ct * 131072);
#pragma unroll
    for (int ct = 0; ct < 4; ++ct)
      acc[ct] = wmma16(a_cur.v, b_cur[ct], acc[ct]);
    a_cur = a_nxt;
#pragma unroll
    for (int ct = 0; ct < 4; ++ct) b_cur[ct] = b_nxt[ct];
  }
#pragma unroll
  for (int ct = 0; ct < 4; ++ct)
    acc[ct] = wmma16(a_cur.v, b_cur[ct], acc[ct]);

#pragma unroll
  for (int ct = 0; ct < 4; ++ct)
#pragma unroll
    for (int r = 0; r < 8; ++r) {
      int orow = rowt * 16 + r + laneHi * 8;
      int col  = wv * 64 + ct * 16 + (lane & 15);
      partial[((size_t)kch * ROWS + orow) * NCOEF + col] = acc[ct][r];
    }
}

// --------------------------------------------------- reduce + complex mix ---
// grid (64 mode-slots, 32 batches), block 64
__global__ void k_mix(const float* __restrict__ partial,
                      const float* __restrict__ wp, const float* __restrict__ wn,
                      const float* __restrict__ phi_p, const float* __restrict__ phi_n,
                      _Float16* __restrict__ afused) {
  __shared__ float sXr[64], sXi[64];
  int s = blockIdx.x, b = blockIdx.y;
  int i = threadIdx.x;
  int mi = (s < 32) ? s : (s - 32);
  int recol = (s < 32) ? s : (64 + mi);
  int imcol = recol + 32;
  const float* w   = (s < 32) ? wp : wn;
  const float* phi = ((s < 32) ? phi_p : phi_n) + (b * 32 + mi) * 2;

  float xr = 0, xi = 0;
  for (int ch = 0; ch < KCH; ++ch) {
    size_t base = ((size_t)ch * ROWS + b * 64 + i) * NCOEF;
    xr += partial[base + recol];
    xi += partial[base + imcol];
  }
  sXr[i] = xr; sXi[i] = xi;
  __syncthreads();

  int o = threadIdx.x;
  float ar = 0, ai = 0;
  for (int ii = 0; ii < 64; ++ii) {
    const float* wpt = w + (((size_t)ii * 64 + o) * 32 + mi) * 2;
    float wr = wpt[0], wi = wpt[1];
    ar += sXr[ii] * wr - sXi[ii] * wi;
    ai += sXr[ii] * wi + sXi[ii] * wr;
  }
  float pr = phi[0], pi = phi[1];
  _Float16* arow = afused + (size_t)(b * 64 + o) * KF;
  arow[recol] = (_Float16)(ar * pr - ai * pi);
  arow[imcol] = (_Float16)(ar * pi + ai * pr);
}

// ------------------------------------------------------- linear part of A ---
__global__ void k_lin(const float* __restrict__ gamma, const float* __restrict__ beta,
                      const float* __restrict__ lin_w, const float* __restrict__ lin_b,
                      _Float16* __restrict__ afused, float* __restrict__ bias) {
  int b = blockIdx.x, o = threadIdx.x;
  float g = 1.0f + gamma[b * 64 + o];
  _Float16* arow = afused + (size_t)(b * 64 + o) * KF + NCOEF;
  for (int c = 0; c < 64; ++c) arow[c] = (_Float16)(g * lin_w[o * 64 + c]);
  bias[b * 64 + o] = g * lin_b[o] + beta[b * 64 + o];
}

// ------------------------------------------- fused synthesis + linear GEMM --
// grid (256 ntile-pairs, 32 batches), block 64 (2 waves; wave handles one ntile)
// Double-buffered across the 6 K-steps.
__global__ void k_final(const _Float16* __restrict__ afused,
                        const _Float16* __restrict__ gfrag,
                        const _Float16* __restrict__ x16,
                        const float* __restrict__ bias,
                        float* __restrict__ out) {
  int lane = threadIdx.x & 31;
  int wv   = threadIdx.x >> 5;
  int laneHi = lane >> 4;
  int nt = blockIdx.x * 2 + wv;
  int b  = blockIdx.y;

  __shared__ float sB[64];
  if (threadIdx.x < 64) sB[threadIdx.x] = bias[b * 64 + threadIdx.x];
  __syncthreads();

  const _Float16* Ab = afused + (size_t)b * 64 * KF + (size_t)(lane & 15) * KF;
  const _Float16* gb = gfrag + ((size_t)nt * 4 * 32 + lane) * 16;
  const _Float16* xb = x16 + (size_t)b * 64 * LL + nt * 16 + (lane & 15);

  v8f acc[4] = {};
  AV bf_cur; AV a_cur[4];

  // preload kk = 0
  bf_cur.v = *(const v16h*)gb;
#pragma unroll
  for (int mt = 0; mt < 4; ++mt) {
    const _Float16* ar = Ab + (size_t)(mt * 16) * KF;
    a_cur[mt].h8[0] = *(const v8h*)(ar + (laneHi << 3));
    a_cur[mt].h8[1] = *(const v8h*)(ar + 16 + (laneHi << 3));
  }

#pragma unroll
  for (int kk = 0; kk < 6; ++kk) {
    AV bf_nxt; AV a_nxt[4];
    if (kk < 5) {
      int kn = kk + 1;
      if (kn < 4) {
        bf_nxt.v = *(const v16h*)(gb + (size_t)kn * 32 * 16);
      } else {
        int c0 = (kn - 4) * 32;
#pragma unroll
        for (int h = 0; h < 16; ++h) {
          int c = c0 + kmap(h, laneHi);
          bf_nxt.v[h] = xb[(size_t)c * LL];
        }
      }
#pragma unroll
      for (int mt = 0; mt < 4; ++mt) {
        const _Float16* ar = Ab + (size_t)(mt * 16) * KF + kn * 32;
        a_nxt[mt].h8[0] = *(const v8h*)(ar + (laneHi << 3));
        a_nxt[mt].h8[1] = *(const v8h*)(ar + 16 + (laneHi << 3));
      }
    }
#pragma unroll
    for (int mt = 0; mt < 4; ++mt)
      acc[mt] = wmma16(a_cur[mt].v, bf_cur.v, acc[mt]);
    if (kk < 5) {
      bf_cur = bf_nxt;
#pragma unroll
      for (int mt = 0; mt < 4; ++mt) a_cur[mt] = a_nxt[mt];
    }
  }

#pragma unroll
  for (int mt = 0; mt < 4; ++mt)
#pragma unroll
    for (int r = 0; r < 8; ++r) {
      int o = mt * 16 + laneHi * 8 + r;
      int l = nt * 16 + (lane & 15);
      float v = acc[mt][r] + sB[o];
      out[((size_t)(b * 64 + o)) * LL + l] = v * sigmoidf(v);
    }
}

// ---------------------------------------------------------------------------
extern "C" void kernel_launch(void* const* d_in, const int* in_sizes, int n_in,
                              void* d_out, int out_size, void* d_ws, size_t ws_size,
                              hipStream_t stream) {
  const float* x     = (const float*)d_in[0];
  const float* emb   = (const float*)d_in[1];
  const float* wp    = (const float*)d_in[2];
  const float* wn    = (const float*)d_in[3];
  const float* Arp   = (const float*)d_in[4];
  const float* Aip   = (const float*)d_in[5];
  const float* Arn   = (const float*)d_in[6];
  const float* Ain   = (const float*)d_in[7];
  const float* tw1   = (const float*)d_in[8];
  const float* tb1   = (const float*)d_in[9];
  const float* tw2   = (const float*)d_in[10];
  const float* tb2   = (const float*)d_in[11];
  const float* linw  = (const float*)d_in[12];
  const float* linb  = (const float*)d_in[13];
  float* out = (float*)d_out;

  constexpr size_t SZ_X16   = (size_t)ROWS * LL * 2;            // 32 MiB
  constexpr size_t SZ_FFRAG = (size_t)8 * 256 * 32 * 16 * 2;    // 2 MiB
  constexpr size_t SZ_GFRAG = (size_t)NTILES * 4 * 32 * 16 * 2; // 2 MiB
  constexpr size_t SZ_PART  = (size_t)KCH * ROWS * NCOEF * 4;   // 8 MiB
  constexpr size_t SZ_AF    = (size_t)BB * CC * KF * 2;         // 768 KiB
  constexpr size_t SZ_PHI   = (size_t)BB * MM * 2 * 4;
  constexpr size_t SZ_VEC   = (size_t)BB * CC * 4;

  char* w = (char*)d_ws;
  _Float16* x16   = (_Float16*)(w);
  _Float16* ffrag = (_Float16*)(w + SZ_X16);
  _Float16* gfrag = (_Float16*)(w + SZ_X16 + SZ_FFRAG);
  float*    part  = (float*)   (w + SZ_X16 + SZ_FFRAG + SZ_GFRAG);
  _Float16* af    = (_Float16*)(w + SZ_X16 + SZ_FFRAG + SZ_GFRAG + SZ_PART);
  float*    phip  = (float*)   (w + SZ_X16 + SZ_FFRAG + SZ_GFRAG + SZ_PART + SZ_AF);
  float*    phin  = phip + BB * MM * 2;
  float*    gam   = phin + BB * MM * 2;
  float*    bet   = gam + BB * CC;
  float*    bias  = bet + BB * CC;
  size_t need = SZ_X16 + SZ_FFRAG + SZ_GFRAG + SZ_PART + SZ_AF + 2 * SZ_PHI + 3 * SZ_VEC;
  if (ws_size < need) return;

  k_tables <<<dim3(8192), 256, 0, stream>>>(ffrag, gfrag);
  k_convert<<<dim3(8192), 256, 0, stream>>>(x, x16);
  k_mlp    <<<dim3(32), 64, 0, stream>>>(emb, Arp, Aip, Arn, Ain, tw1, tb1, tw2, tb2,
                                         phip, phin, gam, bet);
  k_dft    <<<dim3(128, 8), 64, 0, stream>>>(x16, ffrag, part);
  k_mix    <<<dim3(64, 32), 64, 0, stream>>>(part, wp, wn, phip, phin, af);
  k_lin    <<<dim3(32), 64, 0, stream>>>(gam, bet, linw, linb, af, bias);
  k_final  <<<dim3(256, 32), 64, 0, stream>>>(af, gfrag, x16, bias, out);
}
